// QLSTMQuantum_65481071395563
// MI455X (gfx1250) — matrix operand order repaired and checked by reading, not compiled, not run on hardware
//
#include <hip/hip_runtime.h>
#include <hip/hip_bf16.h>

// ---------------- problem constants ----------------
#define T_STEPS 256
#define BATCH   2048
#define D_IN    128
#define NQ      8
#define N_COLS  32                       // 4 gates * 8 qubits
#define M_ROWS  (T_STEPS * BATCH)        // 524288
#define M_TILES (M_ROWS / 16)            // 32768
#define TILES_PER_WAVE 8

typedef __attribute__((ext_vector_type(16))) _Float16 v16h;
typedef __attribute__((ext_vector_type(2)))  _Float16 v2h;
typedef __attribute__((ext_vector_type(8)))  float    v8f;
typedef __attribute__((ext_vector_type(4)))  float    v4f;

// ---------------- workspace layout (bytes) ----------------
//  [0, 32MB)   : Zh (T*B*32 f16, column-interleaved) -- L2 resident (192MB L2)
//  [32MB,+8KB) : B WMMA fragments (2 ntiles * 4 ksteps * 32 lanes * 16 f16)
//  then 128B   : bias+theta fused (32 f32)
//  then 1KB    : Wh (4 gates * 8 k * 8 q f32)
#define WS_Z_BYTES   ((size_t)M_ROWS * N_COLS * 2)
#define WS_BFRAG_OFF WS_Z_BYTES
#define WS_BT_OFF    (WS_BFRAG_OFF + 8192)
#define WS_WH_OFF    (WS_BT_OFF + 128)

// ============================================================
// Prep: build f16 B fragments in WMMA lane layout, fused bias+theta,
// and the 8x8 recurrent weight blocks. 1 block, 32 threads (one wave).
// ============================================================
__global__ void qlstm_prep_kernel(const float* __restrict__ Wf, const float* __restrict__ bf,
                                  const float* __restrict__ Wi, const float* __restrict__ bi,
                                  const float* __restrict__ Wg, const float* __restrict__ bg,
                                  const float* __restrict__ Wo, const float* __restrict__ bo,
                                  const float* __restrict__ thf, const float* __restrict__ thi,
                                  const float* __restrict__ thg, const float* __restrict__ tho,
                                  _Float16* __restrict__ wsB, float* __restrict__ wsBT,
                                  float* __restrict__ wsWh) {
    const int L = threadIdx.x;                 // 0..31
    const float* Wp[4] = {Wf, Wi, Wg, Wo};
    const float* bp[4] = {bf, bi, bg, bo};
    const float* tp[4] = {thf, thi, thg, tho};
    const int g = L >> 3, q = L & 7;

    // fused bias + RX angle (cos(lin+theta): fold both constants)
    wsBT[L] = bp[g][q] + tp[g][q];

    // recurrent 8x8 blocks: rows 128..135 of each W  -> wsWh[g*64 + k*8 + q]
    #pragma unroll
    for (int k = 0; k < 8; ++k)
        wsWh[g * 64 + k * 8 + q] = Wp[g][(D_IN + k) * NQ + q];

    // B fragments for v_wmma_f32_16x16x32_f16.
    // B is KxN (32x16): lane L holds column N = L%16, K = (L/16)*8 + e for e<8, 16 + (L/16)*8 + (e-8) for e>=8.
    const int hw = L >> 4, nl = L & 15;
    for (int nt = 0; nt < 2; ++nt) {
        const int ng = nt * 16 + nl;           // global column 0..31
        const int gg = ng >> 3, qq = ng & 7;
        for (int kk = 0; kk < 4; ++kk) {       // K-step of 32
            #pragma unroll
            for (int e = 0; e < 16; ++e) {
                const int kglob = kk * 32 + (e < 8 ? hw * 8 + e : 16 + hw * 8 + (e - 8));
                wsB[(size_t)(nt * 4 + kk) * 512 + L * 16 + e] =
                    (_Float16)Wp[gg][kglob * NQ + qq];
            }
        }
    }
}

// ============================================================
// Phase 1: Zh[M,32] = f16( X[M,128] @ Wcat[128,32] + (bias+theta) )
// Zh column layout is interleaved: value (row, col = nt*16+nl) stored at
// Zh[row*32 + nl*2 + nt] so each lane emits one packed b32 store per row.
// Each wave does 8 consecutive 16-row M tiles; B fragments in registers across tiles.
// 8 v_wmma per tile (2 N-tiles x 4 K-steps of 32).
// ============================================================
__global__ void __launch_bounds__(256)
qlstm_gemm_kernel(const float* __restrict__ X, const _Float16* __restrict__ wsB,
                  const float* __restrict__ wsBT, _Float16* __restrict__ Zh) {
    const int lane = threadIdx.x & 31;
    const int wave = blockIdx.x * (blockDim.x >> 5) + (threadIdx.x >> 5);
    const int hw   = lane >> 4;                // half-wave select
    const int nl   = lane & 15;                // row within tile (A) / column (B,C)

    // B fragments: 2 N-tiles x 4 K-steps, held in VGPRs for all tiles
    v16h bfrag[2][4];
    #pragma unroll
    for (int nt = 0; nt < 2; ++nt)
        #pragma unroll
        for (int kk = 0; kk < 4; ++kk)
            bfrag[nt][kk] = *(const v16h*)(wsB + (size_t)(nt * 4 + kk) * 512 + lane * 16);

    const float bt0 = wsBT[nl];
    const float bt1 = wsBT[16 + nl];

    const int tile0 = wave * TILES_PER_WAVE;   // grid sized so all tiles valid
    for (int t = 0; t < TILES_PER_WAVE; ++t) {
        const int tile = tile0 + t;
        const size_t row = (size_t)tile * 16 + nl;   // A: M = lane%16 within tile
        const float* xr = X + row * D_IN;

        // prefetch next tile's row for this lane (clamped to last valid row)
        {
            size_t nrow = row + 16;
            if (nrow >= (size_t)M_ROWS) nrow = (size_t)M_ROWS - 16 + nl;
            __builtin_prefetch(X + nrow * D_IN, 0, 3);
        }

        v8f acc0, acc1;
        #pragma unroll
        for (int r = 0; r < 8; ++r) { acc0[r] = bt0; acc1[r] = bt1; }

        #pragma unroll
        for (int kk = 0; kk < 4; ++kk) {
            // A fragment: e0..7 -> K = kk*32 + hw*8 + e ; e8..15 -> +16
            const v4f f0 = *(const v4f*)(xr + kk * 32 + hw * 8);
            const v4f f1 = *(const v4f*)(xr + kk * 32 + hw * 8 + 4);
            const v4f f2 = *(const v4f*)(xr + kk * 32 + 16 + hw * 8);
            const v4f f3 = *(const v4f*)(xr + kk * 32 + 16 + hw * 8 + 4);
            v16h a;
            #pragma unroll
            for (int e = 0; e < 4; ++e) {
                a[e]      = (_Float16)f0[e];
                a[4 + e]  = (_Float16)f1[e];
                a[8 + e]  = (_Float16)f2[e];
                a[12 + e] = (_Float16)f3[e];
            }
            acc0 = __builtin_amdgcn_wmma_f32_16x16x32_f16(false, a, false, bfrag[0][kk],
                                                          (short)0, acc0, false, false);
            acc1 = __builtin_amdgcn_wmma_f32_16x16x32_f16(false, a, false, bfrag[1][kk],
                                                          (short)0, acc1, false, false);
        }

        // D layout: VGPR r -> M = hw*8 + r, N = nl (acc0) / 16+nl (acc1).
        // Pack the two columns into one b32 store at interleaved position nl*2.
        _Float16* zr = Zh + (size_t)tile * 16 * N_COLS;
        #pragma unroll
        for (int r = 0; r < 8; ++r) {
            v2h p;
            p[0] = (_Float16)acc0[r];
            p[1] = (_Float16)acc1[r];
            *(v2h*)(zr + (hw * 8 + r) * N_COLS + nl * 2) = p;
        }
    }
}

// ============================================================
// Phase 2: recurrent scan. One wave per batch row.
// lane = gate*8 + q. h,c replicated per 8-lane group (indexed by q).
// Zh is column-interleaved: lane L's column (==L) lives at (L&15)*2 + (L>>4).
// ============================================================
__global__ void __launch_bounds__(256)
qlstm_scan_kernel(const _Float16* __restrict__ Zh, const float* __restrict__ wsWh,
                  float* __restrict__ out) {
    const int lane = threadIdx.x & 31;
    const int b    = blockIdx.x * (blockDim.x >> 5) + (threadIdx.x >> 5);  // 0..BATCH-1
    const int g    = lane >> 3, q = lane & 7;
    const int sidx = (lane & 15) * 2 + (lane >> 4);   // interleaved column slot

    float wh[8];
    #pragma unroll
    for (int k = 0; k < 8; ++k) wh[k] = wsWh[g * 64 + k * 8 + q];

    float h = 0.f, c = 0.f;

    for (int t = 0; t < T_STEPS; ++t) {
        float z = (float)Zh[((size_t)t * BATCH + b) * N_COLS + sidx];
        #pragma unroll
        for (int k = 0; k < 8; ++k)
            z += __shfl(h, k, 32) * wh[k];     // h_k lives (replicated) at lanes q==k

        const float a = __cosf(z);             // quantum RX encode + <Z> readout
        float v;
        if (g == 2) v = tanhf(a);              // candidate gate
        else        v = 1.f / (1.f + __expf(-a));  // f, i, o gates

        const float fv = __shfl(v,      q, 32);
        const float iv = __shfl(v,  8 + q, 32);
        const float gv = __shfl(v, 16 + q, 32);
        const float ov = __shfl(v, 24 + q, 32);

        c = fv * c + iv * gv;
        h = ov * tanhf(c);

        if (lane < 8)
            out[((size_t)t * BATCH + b) * NQ + q] = h;
    }
    if (lane < 8) {
        out[(size_t)T_STEPS * BATCH * NQ + (size_t)b * NQ + q]                      = h;
        out[(size_t)T_STEPS * BATCH * NQ + (size_t)BATCH * NQ + (size_t)b * NQ + q] = c;
    }
}

// ============================================================
extern "C" void kernel_launch(void* const* d_in, const int* in_sizes, int n_in,
                              void* d_out, int out_size, void* d_ws, size_t ws_size,
                              hipStream_t stream) {
    const float* X   = (const float*)d_in[0];
    const float* Wf  = (const float*)d_in[1];
    const float* bf  = (const float*)d_in[2];
    const float* Wi  = (const float*)d_in[3];
    const float* bi  = (const float*)d_in[4];
    const float* Wg  = (const float*)d_in[5];
    const float* bg  = (const float*)d_in[6];
    const float* Wo  = (const float*)d_in[7];
    const float* bo  = (const float*)d_in[8];
    const float* thf = (const float*)d_in[9];
    const float* thi = (const float*)d_in[10];
    const float* thg = (const float*)d_in[11];
    const float* tho = (const float*)d_in[12];

    char* ws = (char*)d_ws;
    _Float16* Zh   = (_Float16*)ws;
    _Float16* wsB  = (_Float16*)(ws + WS_BFRAG_OFF);
    float*    wsBT = (float*)(ws + WS_BT_OFF);
    float*    wsWh = (float*)(ws + WS_WH_OFF);
    float*    out  = (float*)d_out;

    // 1) tiny prep: one wave
    qlstm_prep_kernel<<<1, 32, 0, stream>>>(Wf, bf, Wi, bi, Wg, bg, Wo, bo,
                                            thf, thi, thg, tho, wsB, wsBT, wsWh);

    // 2) big GEMM: 32768 M-tiles / 8 per wave = 4096 waves -> 512 blocks x 256 thr
    const int gemm_blocks = M_TILES / (TILES_PER_WAVE * 8); // 8 waves per block
    qlstm_gemm_kernel<<<gemm_blocks, 256, 0, stream>>>(X, wsB, wsBT, Zh);

    // 3) scan: 2048 waves -> 256 blocks x 256 thr
    qlstm_scan_kernel<<<BATCH / 8, 256, 0, stream>>>(Zh, wsWh, out);
}
